// PointnetSAModuleMSG_WithSampling_62294205661896
// MI455X (gfx1250) — compile-verified, matrix operand order checked by reading, not run on hardware
//
#include <hip/hip_runtime.h>

// ---------------------------------------------------------------------------
// PointNet++ MSG SA module for MI455X (gfx1250, wave32).
// Pipeline: weight-prep(f32->f16, K-pad) -> fps -> ballquery(both radii)
//           -> group+MLP+pool (WMMA f16, f32 acc) x2 -> agg (WMMA)
// All GEMM inner loops: pure f16 b128 loads + v_wmma_f32_16x16x32_f16.
// ---------------------------------------------------------------------------

#define B_      4
#define N_      16384
#define C_IN_   64
#define NPOINT_ 2048

typedef __attribute__((ext_vector_type(16))) _Float16 v16h;
typedef __attribute__((ext_vector_type(8)))  float    v8f;

// K-offset of the packed half-pair held in vgpr-pair v (0..7) for the 16-bit
// 16x32 A/B fragment layout (lanes 0-15: K {0-7,16-23}; lanes 16-31: {8-15,24-31}).
__device__ __forceinline__ int a_k0(int v, int h) {
  return ((v & 4) ? 16 : 0) + h * 8 + 2 * (v & 3);
}

// ---------------------------------------------------------------------------
// 0) Weight prep: f32 [Nn][Kw] -> f16 [Nn][Kpad], zero-padded K tail.
// ---------------------------------------------------------------------------
__global__ __launch_bounds__(256) void convert_pad_kernel(const float* __restrict__ src,
                                                          _Float16* __restrict__ dst,
                                                          int Nn, int Kw, int Kpad) {
  int t = blockIdx.x * 256 + threadIdx.x;
  if (t >= Nn * Kpad) return;
  int n = t / Kpad, k = t - n * Kpad;
  dst[t] = (k < Kw) ? (_Float16)src[n * Kw + k] : (_Float16)0.0f;
}

// ---------------------------------------------------------------------------
// 1) Farthest point sampling: 1 workgroup (1024 thr) per batch, points in regs.
// ---------------------------------------------------------------------------
__global__ __launch_bounds__(1024) void fps_kernel(const float* __restrict__ xyz,
                                                   float* __restrict__ new_xyz) {
  const int b   = blockIdx.x;
  const int tid = threadIdx.x;
  const float* xb = xyz + (size_t)b * N_ * 3;

  float px[16], py[16], pz[16], mind[16];
#pragma unroll
  for (int k = 0; k < 16; ++k) {
    int p = k * 1024 + tid;
    px[k] = xb[p * 3 + 0];
    py[k] = xb[p * 3 + 1];
    pz[k] = xb[p * 3 + 2];
    mind[k] = 1e10f;
  }

  __shared__ float s_px, s_py, s_pz;
  __shared__ float wbd[32];
  __shared__ int   wbi[32];

  if (tid == 0) {
    s_px = xb[0]; s_py = xb[1]; s_pz = xb[2];
    float* nx = new_xyz + (size_t)b * NPOINT_ * 3;
    nx[0] = xb[0]; nx[1] = xb[1]; nx[2] = xb[2];
  }
  __syncthreads();

  const int lane = tid & 31, wave = tid >> 5;
  for (int it = 1; it < NPOINT_; ++it) {
    const float cx = s_px, cy = s_py, cz = s_pz;
    float bd = -1.0f;
    int   bi = 0x7fffffff;
#pragma unroll
    for (int k = 0; k < 16; ++k) {
      float dx = px[k] - cx, dy = py[k] - cy, dz = pz[k] - cz;
      float d  = dx * dx + dy * dy + dz * dz;
      float m  = fminf(mind[k], d);
      mind[k]  = m;
      int  gi  = k * 1024 + tid;
      if (m > bd || (m == bd && gi < bi)) { bd = m; bi = gi; }
    }
    // wave32 argmax (prefer lower index on ties, matching jnp.argmax)
#pragma unroll
    for (int off = 16; off > 0; off >>= 1) {
      float od = __shfl_down(bd, off);
      int   oi = __shfl_down(bi, off);
      if (od > bd || (od == bd && oi < bi)) { bd = od; bi = oi; }
    }
    if (lane == 0) { wbd[wave] = bd; wbi[wave] = bi; }
    __syncthreads();
    if (wave == 0) {
      bd = wbd[lane]; bi = wbi[lane];
#pragma unroll
      for (int off = 16; off > 0; off >>= 1) {
        float od = __shfl_down(bd, off);
        int   oi = __shfl_down(bi, off);
        if (od > bd || (od == bd && oi < bi)) { bd = od; bi = oi; }
      }
      if (lane == 0) {
        float qx = xb[bi * 3 + 0], qy = xb[bi * 3 + 1], qz = xb[bi * 3 + 2];
        s_px = qx; s_py = qy; s_pz = qz;
        float* nx = new_xyz + ((size_t)b * NPOINT_ + it) * 3;
        nx[0] = qx; nx[1] = qy; nx[2] = qz;
      }
    }
    __syncthreads();
  }
}

// ---------------------------------------------------------------------------
// 2) Ball query, both radii in one scan. One wave per query; ballot-prefix
//    append reproduces "first nsample indices in ascending order".
// ---------------------------------------------------------------------------
__global__ __launch_bounds__(256) void ballquery_kernel(const float* __restrict__ xyz,
                                                        const float* __restrict__ new_xyz,
                                                        int* __restrict__ idx0,
                                                        int* __restrict__ idx1) {
  const int lane = threadIdx.x & 31;
  const int wave = threadIdx.x >> 5;
  const int Q = blockIdx.x * 8 + wave;           // global query id
  const int b = Q / NPOINT_;
  const float* xb = xyz + (size_t)b * N_ * 3;
  const float* nq = new_xyz + (size_t)Q * 3;
  const float qx = nq[0], qy = nq[1], qz = nq[2];
  const float q2 = qx * qx + qy * qy + qz * qz;
  const float r0sq = (float)(0.8 * 0.8);
  const float r1sq = (float)(1.6 * 1.6);

  int cnt0 = 0, cnt1 = 0, first0 = -1, first1 = -1;
  int* o0 = idx0 + (size_t)Q * 16;
  int* o1 = idx1 + (size_t)Q * 32;
  const unsigned below = (1u << lane) - 1u;

  for (int base = 0; base < N_ && (cnt0 < 16 || cnt1 < 32); base += 32) {
    int   j  = base + lane;
    float x  = xb[j * 3 + 0], y = xb[j * 3 + 1], z = xb[j * 3 + 2];
    float x2 = x * x + y * y + z * z;
    float d2 = q2 + x2 - 2.0f * (qx * x + qy * y + qz * z);  // match reference math
    unsigned m0 = (unsigned)__ballot(d2 <= r0sq);
    unsigned m1 = (unsigned)__ballot(d2 <= r1sq);
    if (m1) {
      int pos = cnt1 + __popc(m1 & below);
      if (((m1 >> lane) & 1u) && pos < 32) o1[pos] = j;
      if (first1 < 0) first1 = base + (__ffs(m1) - 1);
      cnt1 += __popc(m1); if (cnt1 > 32) cnt1 = 32;
    }
    if (m0) {
      int pos = cnt0 + __popc(m0 & below);
      if (((m0 >> lane) & 1u) && pos < 16) o0[pos] = j;
      if (first0 < 0) first0 = base + (__ffs(m0) - 1);
      cnt0 += __popc(m0); if (cnt0 > 16) cnt0 = 16;
    }
  }
  const int pad0 = first0 < 0 ? 0 : first0;
  const int pad1 = first1 < 0 ? 0 : first1;
  for (int p = cnt0 + lane; p < 16; p += 32) o0[p] = pad0;
  for (int p = cnt1 + lane; p < 32; p += 32) o1[p] = pad1;
}

// ---------------------------------------------------------------------------
// One MLP layer over a wave's 16-row tile using v_wmma_f32_16x16x32_f16.
//   Xin : LDS [16][96] f16 row-major (K zero-padded)
//   Wh  : global f16 [NT*16][KT*32] row-major (pre-padded by convert_pad_kernel)
//   A-fragments hoisted across the N-tile loop (KT <= 3 -> <= 24 VGPRs).
//   FINAL: fuse relu + max-pool over the tile's 16 rows into poolOut[NT*16].
// ---------------------------------------------------------------------------
template <bool FINAL, int KT, int NT>
__device__ __forceinline__ void wmma_layer16(const _Float16* __restrict__ Xin,
                                             const _Float16* __restrict__ Wh,
                                             const float* __restrict__ S,
                                             const float* __restrict__ Bb,
                                             _Float16* __restrict__ Xout,
                                             float* __restrict__ poolOut,
                                             int lane) {
  const int h = lane >> 4, ln = lane & 15;
  v16h afr[KT];
#pragma unroll
  for (int kt = 0; kt < KT; ++kt) {
#pragma unroll
    for (int v = 0; v < 8; ++v) {
      const int ka = kt * 32 + a_k0(v, h);
      afr[kt][2 * v]     = Xin[ln * 96 + ka];
      afr[kt][2 * v + 1] = Xin[ln * 96 + ka + 1];
    }
  }
#pragma unroll
  for (int nt = 0; nt < NT; ++nt) {
    v8f acc = {};
    const int n = nt * 16 + ln;
    const _Float16* wr = Wh + (size_t)n * (KT * 32);
#pragma unroll
    for (int kt = 0; kt < KT; ++kt) {
      v16h bw;
#pragma unroll
      for (int v = 0; v < 8; ++v) {
        const int ka = kt * 32 + a_k0(v, h);
        bw[2 * v]     = wr[ka];
        bw[2 * v + 1] = wr[ka + 1];
      }
      acc = __builtin_amdgcn_wmma_f32_16x16x32_f16(false, afr[kt], false, bw,
                                                   (short)0, acc, false, false);
    }
    const float sc = S[n], bi = Bb[n];
    if (FINAL) {
      float m = 0.0f;  // relu outputs are >= 0
#pragma unroll
      for (int i = 0; i < 8; ++i) m = fmaxf(m, fmaxf(acc[i] * sc + bi, 0.0f));
      m = fmaxf(m, __shfl_xor(m, 16));  // combine M=0..7 with M=8..15
      if (h == 0) poolOut[n] = m;
    } else {
#pragma unroll
      for (int i = 0; i < 8; ++i) {
        const float o = fmaxf(acc[i] * sc + bi, 0.0f);
        Xout[(i + 8 * h) * 96 + n] = (_Float16)o;  // C layout -> next A layout
      }
    }
  }
}

// ---------------------------------------------------------------------------
// 3) Gather + 3-layer MLP + max-pool. 256 thr (8 waves); each wave owns one
//    16-row (query,sample) tile through all layers via per-wave LDS slices.
// ---------------------------------------------------------------------------
template <int NS, int L2>
__global__ __launch_bounds__(256) void group_mlp_kernel(
    const float* __restrict__ xyz, const float* __restrict__ new_xyz,
    const float* __restrict__ feats, const int* __restrict__ idx,
    const _Float16* __restrict__ wh1, const _Float16* __restrict__ wh2,
    const _Float16* __restrict__ wh3,
    const float* __restrict__ s1, const float* __restrict__ b1,
    const float* __restrict__ s2, const float* __restrict__ b2,
    const float* __restrict__ s3, const float* __restrict__ b3,
    _Float16* __restrict__ pooled) {
  constexpr int QB = 128 / NS;  // queries per block (128 rows per block)
  __shared__ _Float16 ldsA[8][16][96];
  __shared__ _Float16 ldsB[8][16][96];
  __shared__ float    ldsP[8][128];

  const int lane  = threadIdx.x & 31;
  const int wave  = threadIdx.x >> 5;
  const int Qbase = blockIdx.x * QB;       // global query base (no batch straddle)
  const int b     = Qbase / NPOINT_;
  const float* xb = xyz   + (size_t)b * N_ * 3;
  const float* fb = feats + (size_t)b * C_IN_ * N_;

  _Float16* A  = &ldsA[wave][0][0];
  _Float16* Bf = &ldsB[wave][0][0];

  // Gather this wave's 16 rows: [dx,dy,dz, f0..f63, 0-pad to 96] as f16.
  for (int r = 0; r < 16; ++r) {
    const int rowl = wave * 16 + r;
    const int Q    = Qbase + rowl / NS;
    const int j    = rowl % NS;
    const int p    = idx[(size_t)Q * NS + j];
    const float* nq = new_xyz + (size_t)Q * 3;
    for (int c = lane; c < 96; c += 32) {
      float v;
      if (c < 3)       v = xb[p * 3 + c] - nq[c];
      else if (c < 67) v = fb[(size_t)(c - 3) * N_ + p];
      else             v = 0.0f;
      A[r * 96 + c] = (_Float16)v;
    }
  }
  // Waves are independent here; intra-wave LDS RAW handled by s_wait_dscnt.
  wmma_layer16<false, 3, 4      >(A,  wh1, s1, b1, Bf, nullptr, lane);   // 96 -> 64
  wmma_layer16<false, 2, L2 / 16>(Bf, wh2, s2, b2, A,  nullptr, lane);   // 64 -> L2
  wmma_layer16<true,  L2 / 32, 8>(A,  wh3, s3, b3, nullptr, &ldsP[wave][0], lane); // L2 -> 128
  __syncthreads();

  // Combine wave partials (scale1 has 2 waves per query); store pooled as f16
  // (numerically identical: agg converts to f16 for WMMA anyway).
  constexpr int WPQ = NS / 16;
  constexpr int ENT = (8 / WPQ) * 128;
  for (int e = threadIdx.x; e < ENT; e += 256) {
    const int qq = e >> 7, c = e & 127;
    float v = ldsP[qq * WPQ][c];
    if (WPQ == 2) v = fmaxf(v, ldsP[qq * 2 + 1][c]);
    pooled[((size_t)(Qbase + qq)) * 128 + c] = (_Float16)v;
  }
}

// ---------------------------------------------------------------------------
// 4) Aggregation: [8192 x 256] x [256 x 256]^T, relu, write [B,256,S].
//    A-fragments loaded directly from global f16 pooled in WMMA A layout.
// ---------------------------------------------------------------------------
__global__ __launch_bounds__(128) void agg_kernel(const _Float16* __restrict__ pooled0,
                                                  const _Float16* __restrict__ pooled1,
                                                  const _Float16* __restrict__ wh,
                                                  const float* __restrict__ s,
                                                  const float* __restrict__ bb,
                                                  float* __restrict__ out) {
  const int lane = threadIdx.x & 31, wave = threadIdx.x >> 5;
  const int h = lane >> 4, ln = lane & 15;
  const int tile  = blockIdx.x * 4 + wave;  // 512 M-tiles of 16 queries
  const int Qbase = tile * 16;
  const int b  = Qbase / NPOINT_;
  const int q0 = Qbase % NPOINT_;
  const size_t rowoff = ((size_t)b * NPOINT_ + (q0 + ln)) * 128;

  v16h afr[8];  // preload all 8 K-tiles (cat = [pooled0 | pooled1])
#pragma unroll
  for (int kt = 0; kt < 8; ++kt) {
#pragma unroll
    for (int v = 0; v < 8; ++v) {
      const int ka = kt * 32 + a_k0(v, h);
      const _Float16* src = (ka < 128) ? (pooled0 + rowoff + ka)
                                       : (pooled1 + rowoff + ka - 128);
      afr[kt][2 * v]     = src[0];
      afr[kt][2 * v + 1] = src[1];
    }
  }

  float* ob = out + (size_t)b * 256 * NPOINT_;
  for (int nt = 0; nt < 16; ++nt) {
    v8f acc = {};
    const int o = nt * 16 + ln;
    const _Float16* wr = wh + (size_t)o * 256;
#pragma unroll
    for (int kt = 0; kt < 8; ++kt) {
      v16h bw;
#pragma unroll
      for (int v = 0; v < 8; ++v) {
        const int ka = kt * 32 + a_k0(v, h);
        bw[2 * v]     = wr[ka];
        bw[2 * v + 1] = wr[ka + 1];
      }
      acc = __builtin_amdgcn_wmma_f32_16x16x32_f16(false, afr[kt], false, bw,
                                                   (short)0, acc, false, false);
    }
    const float sc = s[o], bi = bb[o];
#pragma unroll
    for (int i = 0; i < 8; ++i) {
      const float val = fmaxf(acc[i] * sc + bi, 0.0f);
      const int ss = q0 + 8 * h + i;   // C layout: M = i + 8*h
      ob[(size_t)o * NPOINT_ + ss] = val;
    }
  }
}

// ---------------------------------------------------------------------------
extern "C" void kernel_launch(void* const* d_in, const int* in_sizes, int n_in,
                              void* d_out, int out_size, void* d_ws, size_t ws_size,
                              hipStream_t stream) {
  (void)in_sizes; (void)n_in; (void)out_size; (void)ws_size;
  // dict/pytree order: xyz, features, mlp0_w[3], mlp0_s[3], mlp0_b[3],
  //                    mlp1_w[3], mlp1_s[3], mlp1_b[3], agg_w, agg_s, agg_b
  const float* xyz   = (const float*)d_in[0];
  const float* feats = (const float*)d_in[1];
  const float* m0w0 = (const float*)d_in[2];
  const float* m0w1 = (const float*)d_in[3];
  const float* m0w2 = (const float*)d_in[4];
  const float* m0s0 = (const float*)d_in[5];
  const float* m0s1 = (const float*)d_in[6];
  const float* m0s2 = (const float*)d_in[7];
  const float* m0b0 = (const float*)d_in[8];
  const float* m0b1 = (const float*)d_in[9];
  const float* m0b2 = (const float*)d_in[10];
  const float* m1w0 = (const float*)d_in[11];
  const float* m1w1 = (const float*)d_in[12];
  const float* m1w2 = (const float*)d_in[13];
  const float* m1s0 = (const float*)d_in[14];
  const float* m1s1 = (const float*)d_in[15];
  const float* m1s2 = (const float*)d_in[16];
  const float* m1b0 = (const float*)d_in[17];
  const float* m1b1 = (const float*)d_in[18];
  const float* m1b2 = (const float*)d_in[19];
  const float* aggw = (const float*)d_in[20];
  const float* aggs = (const float*)d_in[21];
  const float* aggb = (const float*)d_in[22];

  float* new_xyz = (float*)d_out;                       // [B,2048,3]
  float* out     = (float*)d_out + B_ * NPOINT_ * 3;    // [B,256,2048]

  // ---- workspace layout -------------------------------------------------
  char* ws = (char*)d_ws;
  size_t off = 0;
  int* idx0 = (int*)(ws + off);             off += (size_t)B_ * NPOINT_ * 16 * 4;
  int* idx1 = (int*)(ws + off);             off += (size_t)B_ * NPOINT_ * 32 * 4;
  _Float16* pooled0 = (_Float16*)(ws + off); off += (size_t)B_ * NPOINT_ * 128 * 2;
  _Float16* pooled1 = (_Float16*)(ws + off); off += (size_t)B_ * NPOINT_ * 128 * 2;
  _Float16* wh_m0w0 = (_Float16*)(ws + off); off += (size_t)64  * 96 * 2;
  _Float16* wh_m0w1 = (_Float16*)(ws + off); off += (size_t)64  * 64 * 2;
  _Float16* wh_m0w2 = (_Float16*)(ws + off); off += (size_t)128 * 64 * 2;
  _Float16* wh_m1w0 = (_Float16*)(ws + off); off += (size_t)64  * 96 * 2;
  _Float16* wh_m1w1 = (_Float16*)(ws + off); off += (size_t)96  * 64 * 2;
  _Float16* wh_m1w2 = (_Float16*)(ws + off); off += (size_t)128 * 96 * 2;
  _Float16* wh_agg  = (_Float16*)(ws + off); off += (size_t)256 * 256 * 2;

  // ---- weight prep (tiny, L2-resident) ----------------------------------
  auto cvt = [&](const float* src, _Float16* dst, int Nn, int Kw, int Kpad) {
    int tot = Nn * Kpad;
    convert_pad_kernel<<<(tot + 255) / 256, 256, 0, stream>>>(src, dst, Nn, Kw, Kpad);
  };
  cvt(m0w0, wh_m0w0, 64, 67, 96);
  cvt(m0w1, wh_m0w1, 64, 64, 64);
  cvt(m0w2, wh_m0w2, 128, 64, 64);
  cvt(m1w0, wh_m1w0, 64, 67, 96);
  cvt(m1w1, wh_m1w1, 96, 64, 64);
  cvt(m1w2, wh_m1w2, 128, 96, 96);
  cvt(aggw, wh_agg, 256, 256, 256);

  // ---- main pipeline ----------------------------------------------------
  fps_kernel<<<B_, 1024, 0, stream>>>(xyz, new_xyz);
  ballquery_kernel<<<(B_ * NPOINT_) / 8, 256, 0, stream>>>(xyz, new_xyz, idx0, idx1);
  group_mlp_kernel<16, 64><<<(B_ * NPOINT_) / 8, 256, 0, stream>>>(
      xyz, new_xyz, feats, idx0, wh_m0w0, wh_m0w1, wh_m0w2,
      m0s0, m0b0, m0s1, m0b1, m0s2, m0b2, pooled0);
  group_mlp_kernel<32, 96><<<(B_ * NPOINT_) / 4, 256, 0, stream>>>(
      xyz, new_xyz, feats, idx1, wh_m1w0, wh_m1w1, wh_m1w2,
      m1s0, m1b0, m1s1, m1b1, m1s2, m1b2, pooled1);
  agg_kernel<<<(B_ * NPOINT_) / 16 / 4, 128, 0, stream>>>(pooled0, pooled1,
                                                          wh_agg, aggs, aggb, out);
}
